// UniVSELoss_28484223107535
// MI455X (gfx1250) — matrix operand order repaired and checked by reading, not compiled, not run on hardware
//
#include <hip/hip_runtime.h>
#include <cstddef>
#include <cstdint>

// Problem constants (match reference)
#define B_  128
#define D_  1024
#define R_  36
#define P_  8
#define N_  16
#define PR_ 4

typedef __attribute__((ext_vector_type(16))) _Float16 v16h;
typedef __attribute__((ext_vector_type(8)))  float    v8f;

// ---------------------------------------------------------------------------
// WMMA helpers: D = A(16x32 f16) * B(32x16 f16) + C(16x16 f32), wave32.
// Per-lane addressing is hoisted by the callers; these just do contiguous
// b128 loads + in-register f32->f16 pack conversion.
// ---------------------------------------------------------------------------

__device__ __forceinline__ v8f wmma_acc(v16h a, v16h b, v8f c) {
    // (neg_a, A, neg_b, B, c_mod, C, reuse_a, reuse_b)
    return __builtin_amdgcn_wmma_f32_16x16x32_f16(false, a, false, b,
                                                  (short)0, c, false, false);
}

// A fragment for this lane: halves 0..7 from p[0..7], halves 8..15 from p[16..23].
// (Caller pre-offsets p by +8 floats for lanes 16..31.)
__device__ __forceinline__ v16h cvt_frag_a(const float* __restrict__ p) {
    const float4* p4 = reinterpret_cast<const float4*>(p);
    float t[16];
    *reinterpret_cast<float4*>(&t[0])  = p4[0];
    *reinterpret_cast<float4*>(&t[4])  = p4[1];
    *reinterpret_cast<float4*>(&t[8])  = p4[4];
    *reinterpret_cast<float4*>(&t[12]) = p4[5];
    v16h a;
#pragma unroll
    for (int i = 0; i < 16; ++i) a[i] = (_Float16)t[i];
    return a;
}

// B fragment for this lane: halves 0..7 from p[0..7], halves 8..15 from p[16..23].
// (Caller pre-offsets p by +8 floats for lanes 16..31.)
__device__ __forceinline__ v16h cvt_frag_b(const float* __restrict__ p) {
    const float4* p4 = reinterpret_cast<const float4*>(p);
    float t[16];
    *reinterpret_cast<float4*>(&t[0])  = p4[0];
    *reinterpret_cast<float4*>(&t[4])  = p4[1];
    *reinterpret_cast<float4*>(&t[8])  = p4[4];
    *reinterpret_cast<float4*>(&t[12]) = p4[5];
    v16h b;
#pragma unroll
    for (int i = 0; i < 16; ++i) b[i] = (_Float16)t[i];
    return b;
}

__device__ __forceinline__ float wave_sum(float v) {
#pragma unroll
    for (int m = 16; m > 0; m >>= 1) v += __shfl_xor(v, m, 32);
    return v;
}
__device__ __forceinline__ float wave_max(float v) {
#pragma unroll
    for (int m = 16; m > 0; m >>= 1) v = fmaxf(v, __shfl_xor(v, m, 32));
    return v;
}

// ---------------------------------------------------------------------------
// Kernel 0: zero the loss accumulators (d_ws is poisoned by the harness).
// ---------------------------------------------------------------------------
__global__ void init_ws_kernel(float* __restrict__ w) {
    if (threadIdx.x < 8) w[threadIdx.x] = 0.f;
}

// ---------------------------------------------------------------------------
// Kernel 1: scores[z] = img @ X[z]^T for z in {sent, comp, rel_samp}.
// One wave per 16x16 output tile, K=1024 in 32 WMMA steps.
// ---------------------------------------------------------------------------
__global__ void __launch_bounds__(32)
scores_kernel(const float* __restrict__ img,  const float* __restrict__ sent,
              const float* __restrict__ comp, const float* __restrict__ rel,
              const int* __restrict__ rel_idx, float* __restrict__ scores) {
    const int z    = blockIdx.y;
    const int mt   = blockIdx.x >> 3;
    const int nt   = blockIdx.x & 7;
    const int lane = threadIdx.x & 31;
    const int m    = lane & 15;
    const int hi   = (lane >> 4) & 1;

    // Per-lane row base pointers (hoisted out of the k-loop).
    const float* arow = img + (size_t)(mt * 16 + m) * D_ + hi * 8;
    const int col = nt * 16 + m;
    const float* brow;
    if (z == 0)      brow = sent + (size_t)col * D_;
    else if (z == 1) brow = comp + (size_t)col * D_;
    else             brow = rel + ((size_t)col * PR_ + rel_idx[col]) * D_;  // gather
    brow += hi * 8;

    v8f acc = {};
#pragma unroll 2
    for (int k0 = 0; k0 < D_; k0 += 32)
        acc = wmma_acc(cvt_frag_a(arow + k0), cvt_frag_b(brow + k0), acc);

    // D layout: lane l<16 -> rows m=i, col l; lane l>=16 -> rows m=i+8, col l-16
    float* S = scores + (size_t)z * B_ * B_;
    const int cc    = nt * 16 + m;
    const int mbase = mt * 16 + hi * 8;
#pragma unroll
    for (int i = 0; i < 8; ++i) S[(size_t)(mbase + i) * B_ + cc] = acc[i];
}

// ---------------------------------------------------------------------------
// Kernel 2: contrastive hinge reduction over each 128x128 score matrix.
// z=0 sent (bidirectional) -> slot4, z=1 comp (bi) -> slot3,
// z=2 rel_samp (unidirectional) -> slot2.
// ---------------------------------------------------------------------------
__global__ void __launch_bounds__(256)
contrastive_reduce_kernel(const float* __restrict__ scores, float* __restrict__ slots) {
    const int z = blockIdx.x;
    const float* S = scores + (size_t)z * B_ * B_;
    __shared__ float diag[B_];
    for (int i = threadIdx.x; i < B_; i += blockDim.x) diag[i] = S[(size_t)i * B_ + i];
    __syncthreads();

    float sum = 0.f;
    for (int idx = threadIdx.x; idx < B_ * B_; idx += blockDim.x) {
        const int i = idx >> 7;
        const int j = idx & (B_ - 1);
        if (i != j) {
            const float v = S[idx];
            sum += fmaxf(v - diag[i], 0.f);            // cost_s
            if (z < 2) sum += fmaxf(v - diag[j], 0.f); // cost_im (bidirectional only)
        }
    }
    sum = wave_sum(sum);
    const int slot = (z == 0) ? 4 : (z == 1) ? 3 : 2;
    if ((threadIdx.x & 31) == 0) atomicAdd(&slots[slot], sum);
}

// ---------------------------------------------------------------------------
// Kernel 3: local_loss. grid = (B, 3): z=0 obj->slot0, z=1/2 attr->slot1.
// 9 waves per block: waves 0..7 own the 16 negatives of p=wave (one 16-row
// A tile; per-column max over rows = max over n); wave 8 does the 8 positive
// rows (index-clamped padding; garbage rows never read). Then softmax-
// weighted hinge per p with wave reductions, float-atomic into the slot.
// ---------------------------------------------------------------------------
__global__ void __launch_bounds__(288)
local_loss_kernel(const float* __restrict__ img_feat,
                  const float* __restrict__ obj,  const float* __restrict__ neg_obj,
                  const float* __restrict__ attr, const float* __restrict__ neg_attr_n,
                  const float* __restrict__ neg_attr_a,
                  float* __restrict__ slots) {
    const int b    = blockIdx.x;
    const int z    = blockIdx.y;
    const int wave = threadIdx.x >> 5;
    const int lane = threadIdx.x & 31;
    const int m    = lane & 15;
    const int hi   = (lane >> 4) & 1;

    const float* pos  = (z == 0) ? obj : attr;
    const float* neg  = (z == 0) ? neg_obj : (z == 1) ? neg_attr_n : neg_attr_a;
    float* slot = (z == 0) ? &slots[0] : &slots[1];

    __shared__ float s_max[P_][48];
    __shared__ float s_sim[P_][48];

    const float* feat = img_feat + (size_t)b * R_ * D_;

    // Per-lane A row pointer (index-clamped padding for the pos tile).
    const float* arow;
    if (wave < 8) arow = neg + ((size_t)b * P_ * N_ + wave * N_ + m) * D_ + hi * 8;
    else          arow = pos + ((size_t)b * P_ + (m < P_ ? m : P_ - 1)) * D_ + hi * 8;

    // Per-lane B row pointers for the 3 column tiles (clamped past r=35;
    // garbage columns are never read by the reductions).
    const float* brow[3];
#pragma unroll
    for (int j = 0; j < 3; ++j) {
        const int r = j * 16 + m;
        brow[j] = feat + (size_t)(r < R_ ? r : R_ - 1) * D_ + hi * 8;
    }

    v8f acc0 = {}, acc1 = {}, acc2 = {};
#pragma unroll 2
    for (int k0 = 0; k0 < D_; k0 += 32) {
        // Prefetch the HBM-bound A stream ~1KB ahead (global_prefetch_b8).
        __builtin_prefetch(arow + k0 + 256, 0, 0);
        const v16h a = cvt_frag_a(arow + k0);
        acc0 = wmma_acc(a, cvt_frag_b(brow[0] + k0), acc0);
        acc1 = wmma_acc(a, cvt_frag_b(brow[1] + k0), acc1);
        acc2 = wmma_acc(a, cvt_frag_b(brow[2] + k0), acc2);
    }

    if (wave < 8) {
        // max over the 16 rows (n) of this tile, per column r
        v8f* accs[3] = { &acc0, &acc1, &acc2 };
#pragma unroll
        for (int j = 0; j < 3; ++j) {
            const v8f& d = *accs[j];
            float mv = d[0];
#pragma unroll
            for (int i = 1; i < 8; ++i) mv = fmaxf(mv, d[i]);
            const float other = __shfl_xor(mv, 16, 32);  // combine m 0..7 with 8..15
            if (lane < 16) s_max[wave][j * 16 + lane] = fmaxf(mv, other);
        }
    } else {
        // positive sims: rows m=0..7 live in lanes 0..15 as d[i]=sims[p=i][col]
        v8f* accs[3] = { &acc0, &acc1, &acc2 };
#pragma unroll
        for (int j = 0; j < 3; ++j) {
            const v8f& d = *accs[j];
            if (lane < 16) {
#pragma unroll
                for (int i = 0; i < 8; ++i) s_sim[i][j * 16 + lane] = d[i];
            }
        }
    }
    __syncthreads();

    if (wave < 8) {
        const int p  = wave;
        const int r1 = lane;          // always < 36
        const int r2 = lane + 32;     // valid iff < 36
        const bool v2 = (r2 < R_);

        const float s1 = s_sim[p][r1];
        const float h1 = fmaxf(s_max[p][r1] - s1, 0.f);
        const float s2 = v2 ? s_sim[p][r2] : -1e30f;
        const float h2 = v2 ? fmaxf(s_max[p][r2] - s2, 0.f) : 0.f;

        const float mx  = wave_max(fmaxf(s1, s2));
        const float e1  = __expf(s1 - mx);
        const float e2  = v2 ? __expf(s2 - mx) : 0.f;
        const float den = wave_sum(e1 + e2);
        const float num = wave_sum(e1 * h1 + e2 * h2);
        if (lane == 0) atomicAdd(slot, num / den);
    }
}

// ---------------------------------------------------------------------------
// Kernel 4: global_loss(img, rel, neg_rel) -> slot2. One block per (b,p);
// 17 waves: wave 0 = positive dot, waves 1..16 = negatives. Exact fp32 dots.
// ---------------------------------------------------------------------------
__global__ void __launch_bounds__(544)
global_rel_kernel(const float* __restrict__ img, const float* __restrict__ rel,
                  const float* __restrict__ neg_rel, float* __restrict__ slots) {
    const int b = blockIdx.x >> 2;
    const int p = blockIdx.x & 3;
    __shared__ float s_img[D_];
    __shared__ float s_val[17];
    for (int i = threadIdx.x; i < D_; i += blockDim.x) s_img[i] = img[(size_t)b * D_ + i];
    __syncthreads();

    const int wave = threadIdx.x >> 5;
    const int lane = threadIdx.x & 31;
    const float* row = (wave == 0)
        ? rel + ((size_t)b * PR_ + p) * D_
        : neg_rel + (((size_t)b * PR_ + p) * N_ + (wave - 1)) * D_;

    float s = 0.f;
    for (int i = lane; i < D_; i += 32) s += row[i] * s_img[i];
    s = wave_sum(s);
    if (lane == 0) s_val[wave] = s;
    __syncthreads();

    if (threadIdx.x == 0) {
        float mx = s_val[1];
#pragma unroll
        for (int n = 2; n <= 16; ++n) mx = fmaxf(mx, s_val[n]);
        atomicAdd(&slots[2], fmaxf(mx - s_val[0], 0.f));
    }
}

// ---------------------------------------------------------------------------
// Kernel 5: finalize. slots = [l_obj, l_attr, l_rel, l_comp, l_sent].
// out = [total, l_obj, l_attr, l_rel, l_comp, l_sent].
// ---------------------------------------------------------------------------
__global__ void finalize_kernel(const float* __restrict__ slots, float* __restrict__ out) {
    if (threadIdx.x == 0) {
        const float l_obj = slots[0], l_attr = slots[1], l_rel = slots[2];
        const float l_comp = slots[3], l_sent = slots[4];
        out[0] = l_sent + 0.5f * l_comp + 0.5f * l_rel + 0.5f * l_attr + 0.5f * l_obj;
        out[1] = l_obj; out[2] = l_attr; out[3] = l_rel; out[4] = l_comp; out[5] = l_sent;
    }
}

// ---------------------------------------------------------------------------
extern "C" void kernel_launch(void* const* d_in, const int* in_sizes, int n_in,
                              void* d_out, int out_size, void* d_ws, size_t ws_size,
                              hipStream_t stream) {
    (void)in_sizes; (void)n_in; (void)out_size; (void)ws_size;
    const float* img        = (const float*)d_in[0];
    const float* sent       = (const float*)d_in[1];
    const float* comp       = (const float*)d_in[2];
    const float* img_feat   = (const float*)d_in[3];
    const float* obj        = (const float*)d_in[4];
    const float* neg_obj    = (const float*)d_in[5];
    const float* attr       = (const float*)d_in[6];
    const float* neg_attr_n = (const float*)d_in[7];
    const float* neg_attr_a = (const float*)d_in[8];
    const float* rel        = (const float*)d_in[9];
    const float* neg_rel    = (const float*)d_in[10];
    const int*   rel_idx    = (const int*)d_in[11];

    float* out    = (float*)d_out;
    float* w      = (float*)d_ws;
    float* slots  = w;        // 5 accumulators (padded to 64 floats)
    float* scores = w + 64;   // 3 x 128x128 fp32 score matrices (192 KB)

    hipLaunchKernelGGL(init_ws_kernel, dim3(1), dim3(32), 0, stream, w);
    hipLaunchKernelGGL(scores_kernel, dim3(64, 3), dim3(32), 0, stream,
                       img, sent, comp, rel, rel_idx, scores);
    hipLaunchKernelGGL(contrastive_reduce_kernel, dim3(3), dim3(256), 0, stream,
                       scores, slots);
    hipLaunchKernelGGL(local_loss_kernel, dim3(B_, 3), dim3(288), 0, stream,
                       img_feat, obj, neg_obj, attr, neg_attr_n, neg_attr_a, slots);
    hipLaunchKernelGGL(global_rel_kernel, dim3(B_ * PR_), dim3(544), 0, stream,
                       img, rel, neg_rel, slots);
    hipLaunchKernelGGL(finalize_kernel, dim3(1), dim3(32), 0, stream, slots, out);
}
